// GRUScratch_6871947673898
// MI455X (gfx1250) — matrix operand order repaired and checked by reading, not compiled, not run on hardware
//
#include <hip/hip_runtime.h>

// Problem constants (from reference)
constexpr int B  = 128;
constexpr int T  = 1024;
constexpr int I  = 256;
constexpr int H  = 256;
constexpr int BT = B * T;          // 131072
constexpr int IH = I + H;          // 512

typedef __attribute__((ext_vector_type(16))) __bf16 bf16x16;
typedef __attribute__((ext_vector_type(8)))  float  f32x8;

union BF16Frag {
    uint4   u4[2];
    bf16x16 v;
};

__device__ __forceinline__ unsigned short f2bf(float f) {
    unsigned int u = __float_as_uint(f);
    unsigned int r = u + 0x7FFFu + ((u >> 16) & 1u);   // round-to-nearest-even
    return (unsigned short)(r >> 16);
}

__device__ __forceinline__ float sigmoid_fast(float x) {
    return 1.0f / (1.0f + __expf(-x));
}

__device__ __forceinline__ float tanh_fast(float x) {
    float e = __expf(-2.0f * x);
    return (1.0f - e) / (1.0f + e);
}

__device__ __forceinline__ f32x8 wmma_bf16(const BF16Frag& a, const BF16Frag& b, f32x8 c) {
    return __builtin_amdgcn_wmma_f32_16x16x32_bf16(
        /*neg_a=*/false, a.v, /*neg_b=*/false, b.v,
        /*c_mod=*/(short)0, c, /*reuse_a=*/false, /*reuse_b=*/false);
}

// ---------------------------------------------------------------------------
// Kernel 1: convert x [B,T,I] f32 -> bf16
// ---------------------------------------------------------------------------
__global__ void cvt_x_kernel(const float* __restrict__ x,
                             unsigned short* __restrict__ xb, int n) {
    int i = blockIdx.x * blockDim.x + threadIdx.x;
    int stride = gridDim.x * blockDim.x;
    for (; i < n; i += stride) xb[i] = f2bf(x[i]);
}

// ---------------------------------------------------------------------------
// Kernel 2: split each W_g[H, I+H] into bf16 Wx[g][H][I] and U[g][H][H]
// ---------------------------------------------------------------------------
__global__ void cvt_w_kernel(const float* __restrict__ Wz,
                             const float* __restrict__ Wr,
                             const float* __restrict__ Wh,
                             unsigned short* __restrict__ wxb,
                             unsigned short* __restrict__ ub) {
    int idx = blockIdx.x * blockDim.x + threadIdx.x;
    int total = 3 * H * IH;
    int stride = gridDim.x * blockDim.x;
    for (; idx < total; idx += stride) {
        int g   = idx / (H * IH);
        int rem = idx % (H * IH);
        int j   = rem / IH;
        int c   = rem % IH;
        const float* W = (g == 0) ? Wz : ((g == 1) ? Wr : Wh);
        unsigned short v = f2bf(W[(size_t)j * IH + c]);
        if (c < I) wxb[((size_t)g * H + j) * I + c]       = v;
        else       ub [((size_t)g * H + j) * H + (c - I)] = v;
    }
}

// ---------------------------------------------------------------------------
// Kernel 3: input projections G[g][m][j] = bias_g[j] + sum_i x[m,i]*Wx[g][j][i]
// One 16x16 output tile per wave; K = 256 -> 8 WMMAs.
// ---------------------------------------------------------------------------
__global__ void __launch_bounds__(256)
proj_kernel(const unsigned short* __restrict__ xb,
            const unsigned short* __restrict__ wxb,
            const float* __restrict__ bz, const float* __restrict__ br,
            const float* __restrict__ bh, float* __restrict__ G) {
    int wave = threadIdx.x >> 5;        // 0..7
    int lane = threadIdx.x & 31;
    int hl   = lane >> 4;               // K-half selector
    int ln   = lane & 15;

    int ntg   = blockIdx.y * 8 + wave;  // 0..47
    int g     = ntg >> 4;               // gate
    int nt    = ntg & 15;               // N tile within gate
    int mtile = blockIdx.x;             // 0..8191

    int arow = mtile * 16 + ln;         // A: M = lane%16
    int j    = nt * 16 + ln;            // B/C: N = lane%16

    const unsigned short* ap = xb  + (size_t)arow * I;
    const unsigned short* bp = wxb + ((size_t)g * H + j) * I;

    f32x8 acc = {};
#pragma unroll
    for (int k0 = 0; k0 < I; k0 += 32) {
        BF16Frag a, b;
        a.u4[0] = *(const uint4*)(ap + k0 + 8 * hl);        // K = k0+8h .. +7
        a.u4[1] = *(const uint4*)(ap + k0 + 16 + 8 * hl);   // K = k0+16+8h .. +7
        b.u4[0] = *(const uint4*)(bp + k0 + 8 * hl);        // K = k0+8h .. +7
        b.u4[1] = *(const uint4*)(bp + k0 + 16 + 8 * hl);   // K = k0+16+8h .. +7
        acc = wmma_bf16(a, b, acc);
    }

    const float* bias = (g == 0) ? bz : ((g == 1) ? br : bh);
    float bb = bias[j];

    float* gout = G + (size_t)g * BT * H;
#pragma unroll
    for (int r = 0; r < 8; r++) {
        int m = mtile * 16 + r + 8 * hl;    // C/D: M = reg + 8*(lane/16)
        gout[(size_t)m * H + j] = acc[r] + bb;
    }
}

// ---------------------------------------------------------------------------
// Kernel 4: sequential recurrence. grid = 8 blocks (16 batch rows each),
// block = 512 threads = 16 waves; wave w owns output columns [16w, 16w+16).
// No cross-workgroup sync: batch rows are independent through the recurrence.
// ---------------------------------------------------------------------------
__global__ void __launch_bounds__(512)
gru_rec_kernel(const float* __restrict__ h0,
               const unsigned char* __restrict__ mask,
               const unsigned short* __restrict__ ub,
               const float* __restrict__ G,
               float* __restrict__ out, float* __restrict__ hlast) {
    __shared__ __align__(16) unsigned short h_bf [16][H];   // 8 KB
    __shared__ __align__(16) unsigned short rh_bf[16][H];   // 8 KB
    __shared__ __align__(16) float          h_f  [16][H];   // 16 KB

    int s    = blockIdx.x;              // batch slice: rows s*16 .. s*16+15
    int tid  = threadIdx.x;
    int wave = tid >> 5;                // 0..15
    int lane = tid & 31;
    int hl   = lane >> 4;
    int ln   = lane & 15;
    int j    = wave * 16 + ln;          // owned output column

    // init hidden state
    for (int idx = tid; idx < 16 * H; idx += 512) {
        int r = idx >> 8, c = idx & (H - 1);
        float v = h0[(size_t)(s * 16 + r) * H + c];
        h_f[r][c]  = v;
        h_bf[r][c] = f2bf(v);
    }
    __syncthreads();

    const unsigned short* Uz = ub;
    const unsigned short* Ur = ub + (size_t)H * H;
    const unsigned short* Uh = ub + (size_t)2 * H * H;
    const float* Gz = G;
    const float* Gr = G + (size_t)BT * H;
    const float* Gh = G + (size_t)2 * BT * H;

    for (int t = 0; t < T; t++) {
        // ---- A fragments of current h (read by all waves) ----
        BF16Frag a[8];
#pragma unroll
        for (int kc = 0; kc < 8; kc++) {
            int k0 = kc * 32;
            a[kc].u4[0] = *(const uint4*)&h_bf[ln][k0 + 8 * hl];
            a[kc].u4[1] = *(const uint4*)&h_bf[ln][k0 + 16 + 8 * hl];
        }

        // ---- z and r pre-activations: init with precomputed x-part ----
        f32x8 accz = {}, accr = {};
#pragma unroll
        for (int r = 0; r < 8; r++) {
            size_t m = (size_t)(s * 16 + r + 8 * hl) * T + t;
            accz[r] = Gz[m * H + j];
            accr[r] = Gr[m * H + j];
            if (t + 1 < T) {                 // prefetch next step's G tiles
                __builtin_prefetch(&Gz[(m + 1) * H + j], 0, 0);
                __builtin_prefetch(&Gr[(m + 1) * H + j], 0, 0);
                __builtin_prefetch(&Gh[(m + 1) * H + j], 0, 0);
            }
        }
#pragma unroll
        for (int kc = 0; kc < 8; kc++) {
            int k0 = kc * 32;
            const unsigned short* pz = Uz + (size_t)j * H + k0 + 8 * hl;
            const unsigned short* pr = Ur + (size_t)j * H + k0 + 8 * hl;
            BF16Frag bz_, br_;
            bz_.u4[0] = *(const uint4*)pz; bz_.u4[1] = *(const uint4*)(pz + 16);
            br_.u4[0] = *(const uint4*)pr; br_.u4[1] = *(const uint4*)(pr + 16);
            accz = wmma_bf16(a[kc], bz_, accz);
            accr = wmma_bf16(a[kc], br_, accr);
        }

        // ---- gates; publish r*h to LDS ----
        float zv[8];
#pragma unroll
        for (int r = 0; r < 8; r++) {
            int lr = r + 8 * hl;
            int b  = s * 16 + lr;
            float z = sigmoid_fast(accz[r]);
            if (mask[(size_t)b * T + t]) z = 1.0f;   // frozen step carries h
            zv[r] = z;
            float rg = sigmoid_fast(accr[r]);
            rh_bf[lr][j] = f2bf(rg * h_f[lr][j]);
        }
        __syncthreads();

        // ---- candidate: Gh + (r*h) @ Uh^T ----
        f32x8 acch = {};
#pragma unroll
        for (int r = 0; r < 8; r++) {
            size_t m = (size_t)(s * 16 + r + 8 * hl) * T + t;
            acch[r] = Gh[m * H + j];
        }
#pragma unroll
        for (int kc = 0; kc < 8; kc++) {
            int k0 = kc * 32;
            BF16Frag af, bh_;
            af.u4[0] = *(const uint4*)&rh_bf[ln][k0 + 8 * hl];
            af.u4[1] = *(const uint4*)&rh_bf[ln][k0 + 16 + 8 * hl];
            const unsigned short* ph = Uh + (size_t)j * H + k0 + 8 * hl;
            bh_.u4[0] = *(const uint4*)ph; bh_.u4[1] = *(const uint4*)(ph + 16);
            acch = wmma_bf16(af, bh_, acch);
        }
        __syncthreads();   // all reads of h_bf / rh_bf for step t are complete

        // ---- blend, emit, update state ----
#pragma unroll
        for (int r = 0; r < 8; r++) {
            int lr = r + 8 * hl;
            int b  = s * 16 + lr;
            float hc = tanh_fast(acch[r]);
            float hn = zv[r] * h_f[lr][j] + (1.0f - zv[r]) * hc;
            out[((size_t)b * T + t) * H + j] = hn;
            h_f[lr][j]  = hn;
            h_bf[lr][j] = f2bf(hn);
        }
        __syncthreads();   // new h visible before next step's A loads
    }

    for (int idx = tid; idx < 16 * H; idx += 512) {
        int r = idx >> 8, c = idx & (H - 1);
        hlast[(size_t)(s * 16 + r) * H + c] = h_f[r][c];
    }
}

// ---------------------------------------------------------------------------
// Host launcher
// ---------------------------------------------------------------------------
extern "C" void kernel_launch(void* const* d_in, const int* in_sizes, int n_in,
                              void* d_out, int out_size, void* d_ws, size_t ws_size,
                              hipStream_t stream) {
    const float*         x    = (const float*)d_in[0];
    const float*         h0   = (const float*)d_in[1];
    const unsigned char* mask = (const unsigned char*)d_in[2];
    const float*         Wz   = (const float*)d_in[3];
    const float*         bz   = (const float*)d_in[4];
    const float*         Wr   = (const float*)d_in[5];
    const float*         br   = (const float*)d_in[6];
    const float*         Wh   = (const float*)d_in[7];
    const float*         bh   = (const float*)d_in[8];

    float* out   = (float*)d_out;
    float* hlast = out + (size_t)B * T * H;

    // workspace layout
    char*  ws     = (char*)d_ws;
    size_t off_xb = 0;                                   // B*T*I bf16 = 64 MB
    size_t off_wx = off_xb + (size_t)B * T * I * 2;      // 3*H*I bf16
    size_t off_u  = off_wx + (size_t)3 * H * I * 2;      // 3*H*H bf16
    size_t off_g  = off_u  + (size_t)3 * H * H * 2;      // 3*BT*H f32 = 384 MB
    unsigned short* xb  = (unsigned short*)(ws + off_xb);
    unsigned short* wxb = (unsigned short*)(ws + off_wx);
    unsigned short* ub  = (unsigned short*)(ws + off_u);
    float*          G   = (float*)(ws + off_g);

    cvt_x_kernel<<<8192, 256, 0, stream>>>(x, xb, B * T * I);
    cvt_w_kernel<<<1536, 256, 0, stream>>>(Wz, Wr, Wh, wxb, ub);

    dim3 pgrid(BT / 16, 6);   // 8192 M-tiles x (48 N-tiles / 8 waves)
    proj_kernel<<<pgrid, 256, 0, stream>>>(xb, wxb, bz, br, bh, G);

    gru_rec_kernel<<<8, 512, 0, stream>>>(h0, mask, ub, G, out, hlast);
}
